// HybridFakeNewsClassifier_71605694759285
// MI455X (gfx1250) — matrix operand order repaired and checked by reading, not compiled, not run on hardware
//
#include <hip/hip_runtime.h>

// ---------------------------------------------------------------------------
// HybridFakeNewsClassifier for MI455X (gfx1250, wave32, WMMA)
//   GCN layer: h = relu( segsum_{edges+selfloops}( (x@W)[src] * dinv[s]*dinv[d] ) + b )
//   head:      sigmoid( relu([bert|pool]@fc1 + b1) @ fc2 + b2 )
// GEMMs: v_wmma_f32_16x16x32_bf16, 16x64 strip per wave (A-fragment reuse,
// 4 independent accumulators for XDL pipelining).
// Edge aggregation: float4 atomic streaming (L2-resident working set).
// ---------------------------------------------------------------------------

#define N_D_FEAT  128
#define N_HIDDEN  128
#define N_BERT    768
#define N_COMB    896
#define N_HALF    448
#define N_MAXENT  20

typedef __bf16 bf16_t;
typedef __attribute__((ext_vector_type(16))) __bf16 v16bf;
typedef __attribute__((ext_vector_type(8)))  __bf16 v8bf;
typedef __attribute__((ext_vector_type(8)))  float  v8f;

__device__ __forceinline__ bf16_t f2bf(float f) {
    union { float f; unsigned int u; } x; x.f = f;
    unsigned int u = x.u;
    unsigned int r = u + 0x7FFFu + ((u >> 16) & 1u);   // round-to-nearest-even
    union { unsigned short s; bf16_t b; } y; y.s = (unsigned short)(r >> 16);
    return y.b;
}
__device__ __forceinline__ float bf2f(bf16_t b) {
    union { unsigned short s; bf16_t b; } y; y.b = b;
    union { float f; unsigned int u; } x; x.u = ((unsigned int)y.s) << 16;
    return x.f;
}

// -------------------------- small utility kernels --------------------------

__global__ void k_fill(float* p, float v, long n) {
    long t = (long)blockIdx.x * blockDim.x + threadIdx.x;
    if (t < n) p[t] = v;
}

__global__ void k_deg_edges(const int* __restrict__ dst, float* __restrict__ deg, long E) {
    long t = (long)blockIdx.x * blockDim.x + threadIdx.x;
    if (t < E) atomicAdd(&deg[dst[t]], 1.0f);
}

__global__ void k_rsqrt(float* p, long n) {
    long t = (long)blockIdx.x * blockDim.x + threadIdx.x;
    if (t < n) p[t] = rsqrtf(p[t]);   // deg >= 1 (self loop) -> always finite
}

__global__ void k_cvt_bf16(const float* __restrict__ in, bf16_t* __restrict__ out, long n) {
    long t = (long)blockIdx.x * blockDim.x + threadIdx.x;
    if (t < n) out[t] = f2bf(in[t]);
}

// Wt[n*K + k] = W[k*N + n]  (fp32 [K,N] -> bf16 [N,K])
__global__ void k_transpose_bf16(const float* __restrict__ W, bf16_t* __restrict__ Wt,
                                 int K, int N) {
    long t = (long)blockIdx.x * blockDim.x + threadIdx.x;
    if (t >= (long)K * N) return;
    int n = (int)(t / K), k = (int)(t % K);
    Wt[t] = f2bf(W[(size_t)k * N + n]);
}

// agg[i,:] = gemm[i,:] * dinv[i]^2   (self-loop term, also zero-initializes agg)
__global__ void k_selfloop_init(const float* __restrict__ g, const float* __restrict__ dinv,
                                float* __restrict__ agg, long n /* nodes*128 */) {
    long t = (long)blockIdx.x * blockDim.x + threadIdx.x;
    if (t >= n) return;
    float di = dinv[t >> 7];
    agg[t] = g[t] * di * di;
}

// one thread per (edge, float4-chunk): agg[dst] += h[src] * dinv[src]*dinv[dst]
__global__ void k_edge_agg(const int* __restrict__ src, const int* __restrict__ dst,
                           const float* __restrict__ dinv, const float* __restrict__ h,
                           float* __restrict__ agg, long E) {
    long t = (long)blockIdx.x * blockDim.x + threadIdx.x;
    long e = t >> 5;
    if (e >= E) return;
    int c = (int)(t & 31);
    int s = src[e], d = dst[e];
    float w = dinv[s] * dinv[d];
    const float4 v = ((const float4*)(h + (size_t)s * N_HIDDEN))[c];
    float* o = agg + (size_t)d * N_HIDDEN + c * 4;
    atomicAdd(o + 0, v.x * w);
    atomicAdd(o + 1, v.y * w);
    atomicAdd(o + 2, v.z * w);
    atomicAdd(o + 3, v.w * w);
}

// out_bf16 = relu(agg + bias[feat]), feat = t & 127
__global__ void k_relu_cvt(const float* __restrict__ agg, const float* __restrict__ b,
                           bf16_t* __restrict__ out, long n) {
    long t = (long)blockIdx.x * blockDim.x + threadIdx.x;
    if (t >= n) return;
    float v = agg[t] + b[t & (N_HIDDEN - 1)];
    out[t] = f2bf(fmaxf(v, 0.0f));
}

// bert fp32 [B,768] -> bf16 cols [0,768) of combined [B,896]
__global__ void k_bert_to_comb(const float* __restrict__ bert, bf16_t* __restrict__ comb,
                               long n /* batch*768 */) {
    long t = (long)blockIdx.x * blockDim.x + threadIdx.x;
    if (t >= n) return;
    long row = t / N_BERT, col = t % N_BERT;
    comb[row * N_COMB + col] = f2bf(bert[t]);
}

// ragged mean over entity ids (-1 padded) -> cols [768,896) of combined
__global__ void k_pool(const bf16_t* __restrict__ h, const int* __restrict__ ent,
                       bf16_t* __restrict__ comb) {
    __shared__ int sid[N_MAXENT];
    const int row = blockIdx.x;
    const int f = threadIdx.x;           // 0..127
    if (f < N_MAXENT) sid[f] = ent[(size_t)row * N_MAXENT + f];
    __syncthreads();
    float acc = 0.0f;
    int cnt = 0;
    #pragma unroll
    for (int m = 0; m < N_MAXENT; ++m) {
        int id = sid[m];
        if (id >= 0) { acc += bf2f(h[(size_t)id * N_HIDDEN + f]); ++cnt; }
    }
    float v = acc / fmaxf((float)cnt, 1.0f);
    comb[(size_t)row * N_COMB + N_BERT + f] = f2bf(v);
}

// logits + sigmoid: out[row] = sigmoid( z[row,:] . w + b )
__global__ void k_fc2_sigmoid(const float* __restrict__ z, const float* __restrict__ w,
                              const float* __restrict__ b, float* __restrict__ out,
                              int batch) {
    int row = blockIdx.x * blockDim.x + threadIdx.x;
    if (row >= batch) return;
    const float* zr = z + (size_t)row * N_HALF;
    float acc = b[0];
    #pragma unroll 4
    for (int k = 0; k < N_HALF; ++k) acc += zr[k] * w[k];
    out[row] = 1.0f / (1.0f + __expf(-acc));
}

// ------------------------------ WMMA GEMM ----------------------------------
// C[M,N] = A[M,K] @ B[K,N] (+bias, relu); A bf16 row-major, Bt = B^T bf16 [N,K].
// One wave computes a 16x64 strip: the A fragment is loaded once per K-step
// and reused by 4 WMMAs into 4 independent accumulators (no D->C chaining
// between consecutive v_wmma -> matrix pipe stays saturated).
// Fragment layouts used here (A and B share a contiguous k order):
//   A (16x32 bf16): lane(row=l&15, kh=l>>4) holds K in [kh*16, kh*16+16)
//   B (32x16 bf16): lane(col=l&15, kh=l>>4) holds K in [kh*16, kh*16+16)
//   C (16x16 f32) : vgpr r -> (M = r + (l>>4)*8, N = l&15)
__global__ void k_gemm_bf16(const bf16_t* __restrict__ A, const bf16_t* __restrict__ Bt,
                            float* __restrict__ C, int M, int N, int K,
                            const float* __restrict__ bias, int relu) {
    const int lane  = threadIdx.x & 31;
    const int wave  = ((int)blockIdx.x * (int)blockDim.x + (int)threadIdx.x) >> 5;
    const int strips = N >> 6;                        // 64-wide strips
    const int tm = wave / strips;
    const int tg = wave % strips;
    if (tm * 16 >= M) return;                         // wave-uniform: EXEC stays all-1s

    const int row = lane & 15;
    const int kh  = lane >> 4;
    int arow = tm * 16 + row; if (arow >= M) arow = M - 1;
    const bf16_t* aRow = A + (size_t)arow * K;
    const bf16_t* bRow0 = Bt + (size_t)(tg * 64 + row) * K;   // row == B column
    const size_t bStep = (size_t)16 * K;                      // next 16-column tile

    v8f c0 = {}, c1 = {}, c2 = {}, c3 = {};
    for (int k0 = 0; k0 < K; k0 += 32) {
        v8bf a0 = *(const v8bf*)(aRow + k0 + kh * 16);
        v8bf a1 = *(const v8bf*)(aRow + k0 + kh * 16 + 8);
        v16bf a;
        #pragma unroll
        for (int i = 0; i < 8; ++i) { a[i] = a0[i]; a[i + 8] = a1[i]; }

        const bf16_t* bk = bRow0 + k0 + kh * 16;
        v8bf p0 = *(const v8bf*)(bk);
        v8bf p1 = *(const v8bf*)(bk + 8);
        v8bf q0 = *(const v8bf*)(bk + bStep);
        v8bf q1 = *(const v8bf*)(bk + bStep + 8);
        v8bf r0 = *(const v8bf*)(bk + 2 * bStep);
        v8bf r1 = *(const v8bf*)(bk + 2 * bStep + 8);
        v8bf s0 = *(const v8bf*)(bk + 3 * bStep);
        v8bf s1 = *(const v8bf*)(bk + 3 * bStep + 8);
        v16bf b0v, b1v, b2v, b3v;
        #pragma unroll
        for (int i = 0; i < 8; ++i) {
            b0v[i] = p0[i]; b0v[i + 8] = p1[i];
            b1v[i] = q0[i]; b1v[i + 8] = q1[i];
            b2v[i] = r0[i]; b2v[i + 8] = r1[i];
            b3v[i] = s0[i]; b3v[i + 8] = s1[i];
        }
        c0 = __builtin_amdgcn_wmma_f32_16x16x32_bf16(false, a, false, b0v, (short)0, c0, false, false);
        c1 = __builtin_amdgcn_wmma_f32_16x16x32_bf16(false, a, false, b1v, (short)0, c1, false, false);
        c2 = __builtin_amdgcn_wmma_f32_16x16x32_bf16(false, a, false, b2v, (short)0, c2, false, false);
        c3 = __builtin_amdgcn_wmma_f32_16x16x32_bf16(false, a, false, b3v, (short)0, c3, false, false);
    }

    const int mbase = tm * 16 + (lane >> 4) * 8;
    const int nlane = lane & 15;
    v8f acc[4] = { c0, c1, c2, c3 };
    #pragma unroll
    for (int j = 0; j < 4; ++j) {
        const int n = tg * 64 + j * 16 + nlane;
        const float bv = bias ? bias[n] : 0.0f;
        #pragma unroll
        for (int r = 0; r < 8; ++r) {
            int m = mbase + r;
            if (m < M) {
                float v = acc[j][r] + bv;
                if (relu) v = fmaxf(v, 0.0f);
                C[(size_t)m * N + n] = v;
            }
        }
    }
}

// ------------------------------- launcher ----------------------------------

static inline unsigned gblocks(long n, int bs) { return (unsigned)((n + bs - 1) / bs); }

extern "C" void kernel_launch(void* const* d_in, const int* in_sizes, int n_in,
                              void* d_out, int out_size, void* d_ws, size_t ws_size,
                              hipStream_t stream) {
    (void)n_in; (void)out_size; (void)ws_size;
    const float* bert  = (const float*)d_in[0];
    const float* x     = (const float*)d_in[1];
    const int*   edges = (const int*)  d_in[2];   // [2,E]: src then dst
    const int*   ent   = (const int*)  d_in[3];
    const float* W1    = (const float*)d_in[4];
    const float* b1    = (const float*)d_in[5];
    const float* W2    = (const float*)d_in[6];
    const float* b2    = (const float*)d_in[7];
    const float* fc1w  = (const float*)d_in[8];
    const float* fc1b  = (const float*)d_in[9];
    const float* fc2w  = (const float*)d_in[10];
    const float* fc2b  = (const float*)d_in[11];
    float* out = (float*)d_out;

    const int  nNodes = in_sizes[1] / N_D_FEAT;
    const long E      = (long)in_sizes[2] / 2;
    const int  batch  = in_sizes[0] / N_BERT;
    const long NH     = (long)nNodes * N_HIDDEN;

    // workspace carve (256B aligned regions)
    char* p = (char*)d_ws;
    auto carve = [&](size_t bytes) -> void* {
        void* r = (void*)p;
        p += (bytes + 255) & ~(size_t)255;
        return r;
    };
    float*  deg   = (float*) carve((size_t)nNodes * 4);            // deg -> dinv in-place
    bf16_t* xb    = (bf16_t*)carve((size_t)NH * 2);                // x / h1 / h2 in bf16
    bf16_t* w1t   = (bf16_t*)carve((size_t)N_D_FEAT * N_HIDDEN * 2);
    bf16_t* w2t   = (bf16_t*)carve((size_t)N_HIDDEN * N_HIDDEN * 2);
    float*  gemm  = (float*) carve((size_t)NH * 4);                // x@W
    float*  agg   = (float*) carve((size_t)NH * 4);                // segment sum
    bf16_t* comb  = (bf16_t*)carve((size_t)batch * N_COMB * 2);    // [bert | pooled]
    bf16_t* fc1t  = (bf16_t*)carve((size_t)N_HALF * N_COMB * 2);   // fc1_w^T
    float*  z     = (float*) carve((size_t)batch * N_HALF * 4);    // relu(fc1)

    const int* srcI = edges;
    const int* dstI = edges + E;

    // ---- GCN symmetric normalization (self loops folded analytically) ----
    k_fill<<<gblocks(nNodes, 256), 256, 0, stream>>>(deg, 1.0f, nNodes); // self loop
    k_deg_edges<<<gblocks(E, 256), 256, 0, stream>>>(dstI, deg, E);
    k_rsqrt<<<gblocks(nNodes, 256), 256, 0, stream>>>(deg, nNodes);      // deg -> dinv

    // ---- operand prep (bf16 + weight transposes) ----
    k_cvt_bf16<<<gblocks(NH, 256), 256, 0, stream>>>(x, xb, NH);
    k_transpose_bf16<<<gblocks((long)N_D_FEAT * N_HIDDEN, 256), 256, 0, stream>>>(W1, w1t, N_D_FEAT, N_HIDDEN);
    k_transpose_bf16<<<gblocks((long)N_HIDDEN * N_HIDDEN, 256), 256, 0, stream>>>(W2, w2t, N_HIDDEN, N_HIDDEN);
    k_transpose_bf16<<<gblocks((long)N_COMB * N_HALF, 256), 256, 0, stream>>>(fc1w, fc1t, N_COMB, N_HALF);

    // ---- GCN layer 1 ----
    {
        long waves = ((long)(nNodes + 15) / 16) * (N_HIDDEN / 64);
        k_gemm_bf16<<<gblocks(waves * 32, 256), 256, 0, stream>>>(
            xb, w1t, gemm, nNodes, N_HIDDEN, N_D_FEAT, nullptr, 0);
    }
    k_selfloop_init<<<gblocks(NH, 256), 256, 0, stream>>>(gemm, deg, agg, NH);
    k_edge_agg<<<gblocks(E * 32, 256), 256, 0, stream>>>(srcI, dstI, deg, gemm, agg, E);
    k_relu_cvt<<<gblocks(NH, 256), 256, 0, stream>>>(agg, b1, xb, NH);   // h1 -> bf16

    // ---- GCN layer 2 ----
    {
        long waves = ((long)(nNodes + 15) / 16) * (N_HIDDEN / 64);
        k_gemm_bf16<<<gblocks(waves * 32, 256), 256, 0, stream>>>(
            xb, w2t, gemm, nNodes, N_HIDDEN, N_HIDDEN, nullptr, 0);
    }
    k_selfloop_init<<<gblocks(NH, 256), 256, 0, stream>>>(gemm, deg, agg, NH);
    k_edge_agg<<<gblocks(E * 32, 256), 256, 0, stream>>>(srcI, dstI, deg, gemm, agg, E);
    k_relu_cvt<<<gblocks(NH, 256), 256, 0, stream>>>(agg, b2, xb, NH);   // h2 -> bf16

    // ---- build combined = [bert | ragged-mean(h2)] in bf16 ----
    k_bert_to_comb<<<gblocks((long)batch * N_BERT, 256), 256, 0, stream>>>(bert, comb, (long)batch * N_BERT);
    k_pool<<<batch, N_HIDDEN, 0, stream>>>(xb, ent, comb);

    // ---- head: relu(comb @ fc1 + b) -> sigmoid(z @ fc2 + b) ----
    {
        long waves = ((long)batch / 16) * (N_HALF / 64);
        k_gemm_bf16<<<gblocks(waves * 32, 256), 256, 0, stream>>>(
            comb, fc1t, z, batch, N_HALF, N_COMB, fc1b, 1);
    }
    k_fc2_sigmoid<<<gblocks(batch, 256), 256, 0, stream>>>(z, fc2w, fc2b, out, batch);
}
